// GNNModel_9371618639896
// MI455X (gfx1250) — compile-verified, matrix operand order checked by reading, not run on hardware
//
#include <hip/hip_runtime.h>

#define NODES   100000
#define EDGES   1600000
#define DIN     128
#define HID     256
#define NCLS    10
#define NGRAPHS 512

static_assert(NODES % 16 == 0, "wave-uniform row guard requires M % 16 == 0");

typedef __attribute__((ext_vector_type(16))) __bf16       v16bf;
typedef __attribute__((ext_vector_type(8)))  float        v8f;
typedef __attribute__((ext_vector_type(4)))  unsigned int u32x4;
typedef __attribute__((ext_vector_type(8)))  unsigned int u32x8;
typedef __attribute__((ext_vector_type(4)))  float        f32x4;

union FragBF { u32x8 u; v16bf v; };

__device__ __forceinline__ unsigned short f2bf(float f) {
    unsigned int u = __float_as_uint(f);
    u += 0x7FFFu + ((u >> 16) & 1u);          // round-to-nearest-even
    return (unsigned short)(u >> 16);
}

__device__ __forceinline__ void load_frag(FragBF& f, const unsigned short* p) {
    const u32x4 lo = *(const u32x4*)p;
    const u32x4 hi = *(const u32x4*)(p + 16);
#pragma unroll
    for (int i = 0; i < 4; ++i) { f.u[i] = lo[i]; f.u[4 + i] = hi[i]; }
}

// ---------------- utility kernels ----------------

__global__ void fill_f32(float* __restrict__ p, float v, int n) {
    int i = blockIdx.x * blockDim.x + threadIdx.x;
    if (i < n) p[i] = v;
}

__global__ void deg_accum(const int* __restrict__ dst, float* __restrict__ deg, int e) {
    int i = blockIdx.x * blockDim.x + threadIdx.x;
    if (i < e) atomicAdd(&deg[dst[i]], 1.0f);
}

__global__ void rsqrt_inplace(float* __restrict__ d, int n) {
    int i = blockIdx.x * blockDim.x + threadIdx.x;
    if (i < n) d[i] = rsqrtf(fmaxf(d[i], 1.0f));
}

__global__ void edge_coeff(const int* __restrict__ src, const int* __restrict__ dst,
                           const float* __restrict__ dinv, float* __restrict__ coeff, int e) {
    int i = blockIdx.x * blockDim.x + threadIdx.x;
    if (i < e) coeff[i] = dinv[src[i]] * dinv[dst[i]];
}

__global__ void conv_f32_bf16(const float* __restrict__ x, unsigned short* __restrict__ o, int n) {
    int i = blockIdx.x * blockDim.x + threadIdx.x;
    if (i < n) o[i] = f2bf(x[i]);
}

// W: K x 256 row-major (f32)  ->  Wt: 256 x K row-major (bf16)
__global__ void transpose_w_bf16(const float* __restrict__ W, unsigned short* __restrict__ Wt, int K) {
    int i = blockIdx.x * blockDim.x + threadIdx.x;
    if (i < 256 * K) {
        int nn = i / K, kk = i - nn * K;
        Wt[(size_t)nn * K + kk] = f2bf(W[(size_t)kk * 256 + nn]);
    }
}

// ---------------- WMMA GEMM: C[M x 256] = A[M x K](bf16) @ Bt^T (Bt is 256 x K bf16) ----------------
// Block = 8 waves; wave w owns rows [m0, m0+16) x cols [n0, n0+128); 8 accumulator tiles.
// B fragments are double-buffered so each WMMA has the next tile's loads in flight.

__global__ __launch_bounds__(256) void gemm_bf16_wmma(
    const unsigned short* __restrict__ A, const unsigned short* __restrict__ Bt,
    float* __restrict__ C, int M, int K)
{
    const int lane = threadIdx.x & 31;
    const int wave = threadIdx.x >> 5;          // 0..7
    const int m0   = blockIdx.x * 128 + wave * 16;
    if (m0 >= M) return;                        // wave-uniform: M % 16 == 0
    const int n0   = blockIdx.y * 128;
    const int mrow = lane & 15;
    const int koff = (lane & 16) ? 8 : 0;       // K-half offset per 16-bit fragment layout

    v8f acc[8];
    v8f z = {};
#pragma unroll
    for (int t = 0; t < 8; ++t) acc[t] = z;

    const unsigned short* arow = A  + (size_t)(m0 + mrow) * K + koff;
    const unsigned short* bcol = Bt + (size_t)(n0 + mrow) * K + koff;  // tile t: + t*16*K
    const size_t bstep = (size_t)16 * K;

    for (int kt = 0; kt < K; kt += 32) {
        FragBF af;
        load_frag(af, arow + kt);

        FragBF bcur, bnext;
        load_frag(bcur, bcol + kt);
#pragma unroll
        for (int t = 0; t < 8; ++t) {
            if (t < 7) load_frag(bnext, bcol + (size_t)(t + 1) * bstep + kt);
            acc[t] = __builtin_amdgcn_wmma_f32_16x16x32_bf16(
                false, af.v, false, bcur.v, (short)0, acc[t], false, false);
            if (t < 7) bcur = bnext;
        }
    }

    const int hi8 = (lane & 16) ? 8 : 0;
#pragma unroll
    for (int t = 0; t < 8; ++t) {
        const int col = n0 + t * 16 + mrow;
#pragma unroll
        for (int i = 0; i < 8; ++i) {
            C[(size_t)(m0 + hi8 + i) * HID + col] = acc[t][i];
        }
    }
}

// ---------------- edge scatter (one wave32 per edge, 8 channels per lane) ----------------

__global__ void scatter_edges(const int* __restrict__ src, const int* __restrict__ dst,
                              const float* __restrict__ coeff, const float* __restrict__ h,
                              float* __restrict__ agg, int E)
{
    int gid  = blockIdx.x * blockDim.x + threadIdx.x;
    int e    = gid >> 5;
    int lane = gid & 31;
    if (e >= E) return;
    int s = src[e], d = dst[e];
    float cf = coeff[e];
    const float* hp = h   + (size_t)s * HID + lane * 8;
    float*       ap = agg + (size_t)d * HID + lane * 8;
    f32x4 v0 = *(const f32x4*)hp;
    f32x4 v1 = *(const f32x4*)(hp + 4);
#pragma unroll
    for (int i = 0; i < 4; ++i) atomicAdd(ap + i, v0[i] * cf);
#pragma unroll
    for (int i = 0; i < 4; ++i) atomicAdd(ap + 4 + i, v1[i] * cf);
}

// ---------------- epilogues: agg + self-loop + bias, ReLU ----------------

__global__ void bias_relu_to_bf16(const float* __restrict__ agg, const float* __restrict__ h,
                                  const float* __restrict__ dinv, const float* __restrict__ b,
                                  unsigned short* __restrict__ out, int total)
{
    int i = blockIdx.x * blockDim.x + threadIdx.x;
    if (i >= total) return;
    int n = i >> 8, c = i & 255;
    float di = dinv[n];
    float v  = agg[i] + h[i] * di * di + b[c];
    out[i] = f2bf(fmaxf(v, 0.0f));
}

__global__ void bias_relu_to_f32(const float* __restrict__ agg, const float* __restrict__ h,
                                 const float* __restrict__ dinv, const float* __restrict__ b,
                                 float* __restrict__ out, int total)
{
    int i = blockIdx.x * blockDim.x + threadIdx.x;
    if (i >= total) return;
    int n = i >> 8, c = i & 255;
    float di = dinv[n];
    float v  = agg[i] + h[i] * di * di + b[c];
    out[i] = fmaxf(v, 0.0f);
}

// ---------------- pooling ----------------

__global__ void pool_accum(const float* __restrict__ h, const int* __restrict__ batch,
                           float* __restrict__ pool, float* __restrict__ cnt, int N)
{
    int gid  = blockIdx.x * blockDim.x + threadIdx.x;
    int n    = gid >> 5;
    int lane = gid & 31;
    if (n >= N) return;
    int g = batch[n];
    const float* hp = h    + (size_t)n * HID + lane * 8;
    float*       pp = pool + (size_t)g * HID + lane * 8;
#pragma unroll
    for (int i = 0; i < 8; ++i) atomicAdd(pp + i, hp[i]);
    if (lane == 0) atomicAdd(&cnt[g], 1.0f);
}

__global__ void pool_div(float* __restrict__ pool, const float* __restrict__ cnt, int total) {
    int i = blockIdx.x * blockDim.x + threadIdx.x;
    if (i >= total) return;
    int g = i >> 8;
    pool[i] = pool[i] / fmaxf(cnt[g], 1.0f);
}

// ---------------- tiny MLP head ----------------

__global__ void fc1_kernel(const float* __restrict__ g, const float* __restrict__ W,
                           const float* __restrict__ b, float* __restrict__ o)
{
    int i = blockIdx.x * blockDim.x + threadIdx.x;   // 512*128
    if (i >= NGRAPHS * 128) return;
    int gg = i >> 7, j = i & 127;
    const float* gr = g + (size_t)gg * HID;
    float s = b[j];
    for (int k = 0; k < HID; ++k) s += gr[k] * W[(size_t)k * 128 + j];
    o[i] = fmaxf(s, 0.0f);
}

__global__ void fc2_kernel(const float* __restrict__ g, const float* __restrict__ W,
                           const float* __restrict__ b, float* __restrict__ o)
{
    int i = blockIdx.x * blockDim.x + threadIdx.x;   // 512*10
    if (i >= NGRAPHS * NCLS) return;
    int gg = i / NCLS, c = i - gg * NCLS;
    const float* gr = g + (size_t)gg * 128;
    float s = b[c];
    for (int k = 0; k < 128; ++k) s += gr[k] * W[(size_t)k * NCLS + c];
    o[i] = s;
}

// ---------------- host launch ----------------

extern "C" void kernel_launch(void* const* d_in, const int* in_sizes, int n_in,
                              void* d_out, int out_size, void* d_ws, size_t ws_size,
                              hipStream_t stream)
{
    const float* x     = (const float*)d_in[0];
    const int*   ei    = (const int*)d_in[1];
    const int*   src   = ei;
    const int*   dst   = ei + EDGES;
    const int*   batch = (const int*)d_in[2];
    const float* W1    = (const float*)d_in[3];
    const float* b1    = (const float*)d_in[4];
    const float* W2    = (const float*)d_in[5];
    const float* b2    = (const float*)d_in[6];
    const float* fw1   = (const float*)d_in[7];
    const float* fb1   = (const float*)d_in[8];
    const float* fw2   = (const float*)d_in[9];
    const float* fb2   = (const float*)d_in[10];
    float* out = (float*)d_out;

    char* ws = (char*)d_ws;
    size_t off = 0;
    auto carve = [&](size_t bytes) -> char* {
        char* p = ws + off;
        off = (off + bytes + 255) & ~(size_t)255;
        return p;
    };
    float*          hbuf = (float*)carve((size_t)NODES * HID * 4);
    float*          agg  = (float*)carve((size_t)NODES * HID * 4);
    unsigned short* abuf = (unsigned short*)carve((size_t)NODES * HID * 2);
    unsigned short* wt   = (unsigned short*)carve((size_t)HID * HID * 2);
    float*          dinv = (float*)carve((size_t)NODES * 4);
    float*          cfb  = (float*)carve((size_t)EDGES * 4);
    float*          pool = (float*)carve((size_t)NGRAPHS * HID * 4);
    float*          cnt  = (float*)carve((size_t)NGRAPHS * 4);
    float*          fc1o = (float*)carve((size_t)NGRAPHS * 128 * 4);

    const int T = 256;
    dim3 gGemm((NODES + 127) / 128, 2, 1);

    // degrees (self-loop folded in as init=1) -> dinv -> per-edge coefficients
    fill_f32<<<(NODES + T - 1) / T, T, 0, stream>>>(dinv, 1.0f, NODES);
    deg_accum<<<(EDGES + T - 1) / T, T, 0, stream>>>(dst, dinv, EDGES);
    rsqrt_inplace<<<(NODES + T - 1) / T, T, 0, stream>>>(dinv, NODES);
    edge_coeff<<<(EDGES + T - 1) / T, T, 0, stream>>>(src, dst, dinv, cfb, EDGES);

    // layer 1
    conv_f32_bf16<<<(NODES * DIN + T - 1) / T, T, 0, stream>>>(x, abuf, NODES * DIN);
    transpose_w_bf16<<<(HID * DIN + T - 1) / T, T, 0, stream>>>(W1, wt, DIN);
    gemm_bf16_wmma<<<gGemm, T, 0, stream>>>(abuf, wt, hbuf, NODES, DIN);
    fill_f32<<<(NODES * HID + T - 1) / T, T, 0, stream>>>(agg, 0.0f, NODES * HID);
    scatter_edges<<<(EDGES * 32 + T - 1) / T, T, 0, stream>>>(src, dst, cfb, hbuf, agg, EDGES);
    bias_relu_to_bf16<<<(NODES * HID + T - 1) / T, T, 0, stream>>>(agg, hbuf, dinv, b1, abuf, NODES * HID);

    // layer 2
    transpose_w_bf16<<<(HID * HID + T - 1) / T, T, 0, stream>>>(W2, wt, HID);
    gemm_bf16_wmma<<<gGemm, T, 0, stream>>>(abuf, wt, hbuf, NODES, HID);
    fill_f32<<<(NODES * HID + T - 1) / T, T, 0, stream>>>(agg, 0.0f, NODES * HID);
    scatter_edges<<<(EDGES * 32 + T - 1) / T, T, 0, stream>>>(src, dst, cfb, hbuf, agg, EDGES);
    bias_relu_to_f32<<<(NODES * HID + T - 1) / T, T, 0, stream>>>(agg, hbuf, dinv, b2, agg, NODES * HID);

    // global mean pool
    fill_f32<<<(NGRAPHS * HID + T - 1) / T, T, 0, stream>>>(pool, 0.0f, NGRAPHS * HID);
    fill_f32<<<(NGRAPHS + T - 1) / T, T, 0, stream>>>(cnt, 0.0f, NGRAPHS);
    pool_accum<<<(NODES * 32 + T - 1) / T, T, 0, stream>>>(agg, batch, pool, cnt, NODES);
    pool_div<<<(NGRAPHS * HID + T - 1) / T, T, 0, stream>>>(pool, cnt, NGRAPHS * HID);

    // MLP head
    fc1_kernel<<<(NGRAPHS * 128 + T - 1) / T, T, 0, stream>>>(pool, fw1, fb1, fc1o);
    fc2_kernel<<<(NGRAPHS * NCLS + T - 1) / T, T, 0, stream>>>(fc1o, fw2, fb2, out);
}
